// PhotonicQuantumWalk_66889820668523
// MI455X (gfx1250) — matrix operand order repaired and checked by reading, not compiled, not run on hardware
//
#include <hip/hip_runtime.h>
#include <hip/hip_bf16.h>

// ---------------------------------------------------------------------------
// PhotonicQuantumWalk on MI455X (gfx1250, wave32)
//
// Pipeline:
//   k_init      : zero accum, state0 = 1/64, zero ampT pad rows
//   k_enc_mask  : fp32 WMMA GEMM adj@enc_w^T (32x32 tile/wave), threshold>0,
//                 store maskT f16 (transposed) -- L2-resident for the walk
//   k_deg       : deg / inv-sqrt-deg from maskT columns
//   64x { k_step1 (coin rotate + fold prev-step norm + amp f16)
//         k_step2 (f16 WMMA maskT@amp shift, 32 rows/wave, iso fixup, loss,
//                  sum s^2 -> accum) }
//   k_probs / k_f1 / k_f2 : readout MLP
// ---------------------------------------------------------------------------

#define BB 8
#define NN 2048
#define STEPS 64

typedef __attribute__((ext_vector_type(2)))  float    v2f;
typedef __attribute__((ext_vector_type(4)))  float    v4f;
typedef __attribute__((ext_vector_type(8)))  float    v8f;
typedef __attribute__((ext_vector_type(8)))  _Float16 v8h;
typedef __attribute__((ext_vector_type(16))) _Float16 v16h;

// ---------------------------------------------------------------------------
__global__ void k_init(float* __restrict__ state0, float* __restrict__ accum,
                       _Float16* __restrict__ ampT) {
  int i = blockIdx.x * blockDim.x + threadIdx.x;          // 65536 threads
  state0[i] = 0.015625f;                                  // 1/sqrt(N*C) = 1/64
  if (i < STEPS * BB) accum[i] = 0.0f;
  if (i < 32768) {                                        // 32768*8 = 262144 halves
    v8h z = {};
    *(v8h*)(ampT + (size_t)i * 8) = z;
  }
}

// ---------------------------------------------------------------------------
// enc GEMM: pre[b][n][k] = sum_m adj[b][n][m]*enc_w[k][m] (+bias), store
// maskT[b][k][n] = (pre>0) as f16 {0,1}.
// fp32 WMMA 16x16x4; one wave = 32x32 tile (4 accumulators, 2x fragment reuse).
__global__ __launch_bounds__(128) void k_enc_mask(
    const float* __restrict__ adj, const float* __restrict__ enc_w,
    const float* __restrict__ enc_b, _Float16* __restrict__ maskT) {
  const int lane = threadIdx.x & 31;
  const int wave = threadIdx.x >> 5;
  const int half = lane >> 4;        // 0: K={0,1}, 1: K={2,3} (32-bit A/B layout)
  const int lm   = lane & 15;
  const int b  = blockIdx.z;
  const int n0 = blockIdx.y * 64 + (wave >> 1) * 32;
  const int k0 = blockIdx.x * 64 + (wave & 1) * 32;

  const float* __restrict__ arow0 = adj + ((size_t)b * NN + (n0 + lm)) * NN;
  const float* __restrict__ arow1 = arow0 + (size_t)16 * NN;
  const float* __restrict__ brow0 = enc_w + (size_t)(k0 + lm) * NN;
  const float* __restrict__ brow1 = brow0 + (size_t)16 * NN;
  const int moff = 2 * half;

  v8f acc00 = {0.f, 0.f, 0.f, 0.f, 0.f, 0.f, 0.f, 0.f};
  v8f acc01 = acc00, acc10 = acc00, acc11 = acc00;

  for (int m = 0; m < NN; m += 16) {
#pragma unroll
    for (int u = 0; u < 4; ++u) {
      const int off = m + 4 * u + moff;
      v2f a0 = *(const v2f*)(arow0 + off);
      v2f a1 = *(const v2f*)(arow1 + off);
      v2f b0 = *(const v2f*)(brow0 + off);
      v2f b1 = *(const v2f*)(brow1 + off);
      acc00 = __builtin_amdgcn_wmma_f32_16x16x4_f32(false, a0, false, b0,
                                                    (short)0, acc00, false, false);
      acc01 = __builtin_amdgcn_wmma_f32_16x16x4_f32(false, a0, false, b1,
                                                    (short)0, acc01, false, false);
      acc10 = __builtin_amdgcn_wmma_f32_16x16x4_f32(false, a1, false, b0,
                                                    (short)0, acc10, false, false);
      acc11 = __builtin_amdgcn_wmma_f32_16x16x4_f32(false, a1, false, b1,
                                                    (short)0, acc11, false, false);
    }
  }

  // D element (r,lane) of subtile (sn,sk): n = n0+16sn + r + 8*half, k = k0+16sk+lm
  const float bias0 = enc_b[k0 + lm];
  const float bias1 = enc_b[k0 + 16 + lm];
  auto emit = [&](const v8f& a, float bias, int k, int nb) {
    v8h mh;
#pragma unroll
    for (int r = 0; r < 8; ++r)
      mh[r] = (_Float16)(((a[r] + bias) > 0.0f) ? 1.0f : 0.0f);
    *(v8h*)(maskT + ((size_t)b * NN + k) * NN + nb + 8 * half) = mh;
  };
  emit(acc00, bias0, k0 + lm,      n0);
  emit(acc01, bias1, k0 + 16 + lm, n0);
  emit(acc10, bias0, k0 + lm,      n0 + 16);
  emit(acc11, bias1, k0 + 16 + lm, n0 + 16);
}

// ---------------------------------------------------------------------------
// deg[b][n] = sum_k maskT[b][k][n];  isd = deg>0 ? 1/sqrt(deg) : 0  (iso <=> isd==0)
__global__ void k_deg(const _Float16* __restrict__ maskT, float* __restrict__ isd) {
  int i = blockIdx.x * blockDim.x + threadIdx.x;  // b*N + n  (16384)
  int b = i >> 11, n = i & (NN - 1);
  const _Float16* base = maskT + (size_t)b * NN * NN + n;
  float d = 0.0f;
  for (int m = 0; m < NN; ++m) d += (float)base[(size_t)m * NN];
  isd[i] = (d > 0.0f) ? (1.0f / sqrtf(d)) : 0.0f;
}

// ---------------------------------------------------------------------------
// coin rotation (normalized complex 2x2) on normalized prev state; write
// st (f32, needed for isolated nodes) and ampT[b][cz][n] = st*isd as f16.
__global__ void k_step1(const float* __restrict__ statePrev,
                        const float* __restrict__ coin_op,
                        const float* __restrict__ isd,
                        const float* __restrict__ accum,
                        float* __restrict__ stOut, _Float16* __restrict__ ampT,
                        int t) {
  int i = blockIdx.x * blockDim.x + threadIdx.x;  // b*N + n  (16384)
  int b = i >> 11, n = i & (NN - 1);

  float c00r = coin_op[0], c00i = coin_op[1], c01r = coin_op[2], c01i = coin_op[3];
  float c10r = coin_op[4], c10i = coin_op[5], c11r = coin_op[6], c11i = coin_op[7];
  float inv = 1.0f / sqrtf(c00r * c00r + c00i * c00i + c01r * c01r + c01i * c01i +
                           c10r * c10r + c10i * c10i + c11r * c11r + c11i * c11i);

  float scale = 1.0f;  // state0 enters un-renormalized, as in the reference
  if (t > 0) scale = 1.0f / (sqrtf(accum[(t - 1) * BB + b]) + 1e-8f);

  v4f s = *(const v4f*)(statePrev + (size_t)i * 4);
  float s0r = s[0] * scale, s0i = s[1] * scale;
  float s1r = s[2] * scale, s1i = s[3] * scale;

  float e0r = inv * (c00r * s0r - c00i * s0i + c01r * s1r - c01i * s1i);
  float e0i = inv * (c00r * s0i + c00i * s0r + c01r * s1i + c01i * s1r);
  float e1r = inv * (c10r * s0r - c10i * s0i + c11r * s1r - c11i * s1i);
  float e1i = inv * (c10r * s0i + c10i * s0r + c11r * s1i + c11i * s1r);

  v4f stv = {e0r, e0i, e1r, e1i};
  *(v4f*)(stOut + (size_t)i * 4) = stv;

  float d = isd[i];
  _Float16* ab = ampT + (size_t)b * 16 * NN + n;   // coalesced across n per row
  ab[0]      = (_Float16)(e0r * d);
  ab[NN]     = (_Float16)(e0i * d);
  ab[2 * NN] = (_Float16)(e1r * d);
  ab[3 * NN] = (_Float16)(e1i * d);
}

// ---------------------------------------------------------------------------
// shifted[b][m][cz] = sum_n maskT[b][m][n]*amp[b][n][cz]; f16 WMMA 16x16x32.
// One wave = 32 m-rows (two tiles sharing each B fragment); maskT is
// L2-resident (67MB < 192MB L2) and read exactly once per step.
__global__ __launch_bounds__(128) void k_step2(
    const _Float16* __restrict__ maskT, const _Float16* __restrict__ ampT,
    const float* __restrict__ stIn, const float* __restrict__ isd,
    float* __restrict__ stateNext, float* __restrict__ accum, int t) {
  const int lane = threadIdx.x & 31;
  const int wave = threadIdx.x >> 5;
  const int half = lane >> 4;
  const int lm   = lane & 15;
  const int b  = blockIdx.y;
  const int m0 = blockIdx.x * 128 + wave * 32;

  const _Float16* __restrict__ Abase0 = maskT + ((size_t)b * NN + (m0 + lm)) * NN;
  const _Float16* __restrict__ Abase1 = Abase0 + (size_t)16 * NN;
  const _Float16* __restrict__ Bbase  = ampT + (size_t)b * 16 * NN + (size_t)lm * NN;

  v8f acc0 = {0.f, 0.f, 0.f, 0.f, 0.f, 0.f, 0.f, 0.f};
  v8f acc1 = acc0;
  for (int n0 = 0; n0 < NN; n0 += 32) {
    __builtin_prefetch(Abase0 + n0 + 256, 0, 0);   // -> global_prefetch_b8
    __builtin_prefetch(Abase1 + n0 + 256, 0, 0);
    // B fragment: lane<16 K=0..15 of column lm, lane>=16 K=16..31 (rows 4..15 zero)
    v16h bf = *(const v16h*)(Bbase + n0 + half * 16);
    // A fragments (contiguous, matching B): lane<16 K=0..15 ; lane>=16 K=16..31
    v8h a00 = *(const v8h*)(Abase0 + n0 + half * 16);
    v8h a01 = *(const v8h*)(Abase0 + n0 + 8 + half * 16);
    v8h a10 = *(const v8h*)(Abase1 + n0 + half * 16);
    v8h a11 = *(const v8h*)(Abase1 + n0 + 8 + half * 16);
    v16h a0, a1;
#pragma unroll
    for (int q = 0; q < 8; ++q) {
      a0[q] = a00[q]; a0[q + 8] = a01[q];
      a1[q] = a10[q]; a1[q + 8] = a11[q];
    }
    acc0 = __builtin_amdgcn_wmma_f32_16x16x32_f16(false, a0, false, bf,
                                                  (short)0, acc0, false, false);
    acc1 = __builtin_amdgcn_wmma_f32_16x16x32_f16(false, a1, false, bf,
                                                  (short)0, acc1, false, false);
  }

  const float lf = __expf(-0.01f * (float)t);
  float sq = 0.0f;
  if (lm < 4) {  // only D columns 0..3 carry cz data
#pragma unroll
    for (int r = 0; r < 8; ++r) {
      {
        int m = m0 + r + 8 * half;
        float im = isd[b * NN + m];
        float v  = (im == 0.0f) ? stIn[((size_t)b * NN + m) * 4 + lm] : acc0[r];
        float sv = lf * v;
        stateNext[((size_t)b * NN + m) * 4 + lm] = sv;
        sq += sv * sv;
      }
      {
        int m = m0 + 16 + r + 8 * half;
        float im = isd[b * NN + m];
        float v  = (im == 0.0f) ? stIn[((size_t)b * NN + m) * 4 + lm] : acc1[r];
        float sv = lf * v;
        stateNext[((size_t)b * NN + m) * 4 + lm] = sv;
        sq += sv * sv;
      }
    }
  }
  for (int off = 16; off > 0; off >>= 1) sq += __shfl_xor(sq, off, 32);
  if (lane == 0) atomicAdd(&accum[t * BB + b], sq);
}

// ---------------------------------------------------------------------------
__global__ void k_probs(const float* __restrict__ stateF,
                        const float* __restrict__ accum, float* __restrict__ probs) {
  int i = blockIdx.x * blockDim.x + threadIdx.x;  // b*N + n
  int b = i >> 11, n = i & (NN - 1);
  float scale = 1.0f / (sqrtf(accum[(STEPS - 1) * BB + b]) + 1e-8f);
  float s2 = scale * scale;
  v4f s = *(const v4f*)(stateF + (size_t)i * 4);
  probs[(size_t)b * 4096 + 2 * n]     = (s[0] * s[0] + s[1] * s[1]) * s2;
  probs[(size_t)b * 4096 + 2 * n + 1] = (s[2] * s[2] + s[3] * s[3]) * s2;
}

__global__ __launch_bounds__(128) void k_f1(const float* __restrict__ probs,
                                            const float* __restrict__ w,
                                            const float* __restrict__ bias,
                                            float* __restrict__ h) {
  int b = blockIdx.x, j = threadIdx.x;
  const float* p  = probs + (size_t)b * 4096;
  const float* wr = w + (size_t)j * 4096;
  float acc = bias[j];
  for (int i = 0; i < 4096; ++i) acc = fmaf(p[i], wr[i], acc);
  h[b * 128 + j] = fmaxf(acc, 0.0f);
}

__global__ __launch_bounds__(64) void k_f2(const float* __restrict__ h,
                                           const float* __restrict__ w,
                                           const float* __restrict__ bias,
                                           float* __restrict__ out) {
  int b = blockIdx.x, j = threadIdx.x;
  const float* hr = h + (size_t)b * 128;
  const float* wr = w + (size_t)j * 128;
  float acc = bias[j];
  for (int i = 0; i < 128; ++i) acc = fmaf(hr[i], wr[i], acc);
  out[b * 64 + j] = acc;
}

// ---------------------------------------------------------------------------
extern "C" void kernel_launch(void* const* d_in, const int* in_sizes, int n_in,
                              void* d_out, int out_size, void* d_ws, size_t ws_size,
                              hipStream_t stream) {
  const float* adj   = (const float*)d_in[0];
  const float* coin  = (const float*)d_in[1];
  const float* enc_w = (const float*)d_in[2];
  const float* enc_b = (const float*)d_in[3];
  const float* f1_w  = (const float*)d_in[4];
  const float* f1_b  = (const float*)d_in[5];
  const float* f2_w  = (const float*)d_in[6];
  const float* f2_b  = (const float*)d_in[7];
  float* out = (float*)d_out;

  char* ws = (char*)d_ws;
  _Float16* maskT = (_Float16*)ws; ws += (size_t)BB * NN * NN * 2;  // 64 MB, L2-resident
  float* stateA   = (float*)ws;    ws += (size_t)BB * NN * 4 * 4;
  float* stateB   = (float*)ws;    ws += (size_t)BB * NN * 4 * 4;
  float* stbuf    = (float*)ws;    ws += (size_t)BB * NN * 4 * 4;
  _Float16* ampT  = (_Float16*)ws; ws += (size_t)BB * 16 * NN * 2;  // 16 rows, 4..15 zero
  float* isd      = (float*)ws;    ws += (size_t)BB * NN * 4;
  float* accum    = (float*)ws;    ws += (size_t)STEPS * BB * 4;
  float* probs    = (float*)ws;    ws += (size_t)BB * 4096 * 4;
  float* h        = (float*)ws;    ws += (size_t)BB * 128 * 4;

  k_init<<<65536 / 256, 256, 0, stream>>>(stateA, accum, ampT);
  k_enc_mask<<<dim3(NN / 64, NN / 64, BB), 128, 0, stream>>>(adj, enc_w, enc_b, maskT);
  k_deg<<<(BB * NN) / 256, 256, 0, stream>>>(maskT, isd);

  float* prev = stateA;
  float* next = stateB;
  for (int t = 0; t < STEPS; ++t) {
    k_step1<<<(BB * NN) / 256, 256, 0, stream>>>(prev, coin, isd, accum, stbuf, ampT, t);
    k_step2<<<dim3(NN / 128, BB), 128, 0, stream>>>(maskT, ampT, stbuf, isd, next, accum, t);
    float* tmp = prev; prev = next; next = tmp;
  }

  k_probs<<<(BB * NN) / 256, 256, 0, stream>>>(prev, accum, probs);
  k_f1<<<BB, 128, 0, stream>>>(probs, f1_w, f1_b, h);
  k_f2<<<BB, 64, 0, stream>>>(h, f2_w, f2_b, out);
}
